// QLoRADense_89378269429979
// MI455X (gfx1250) — matrix operand order repaired and checked by reading, not compiled, not run on hardware
//
#include <hip/hip_runtime.h>
#include <hip/hip_bf16.h>
#include <stdint.h>

// ---------------------------------------------------------------------------
// QLoRA dense: out = bf16( x_bf16 @ dequant_nf4(qweight)*scale + bias ) +
//              (x @ lora_a) @ lora_b * (ALPHA/R)           [fp32 output]
// B=4, S=2048 -> M = 8192 ; K = D_IN = 4096 ; N = D_OUT = 4096
// ---------------------------------------------------------------------------

typedef __attribute__((ext_vector_type(16))) __bf16 bf16x16;
typedef __attribute__((ext_vector_type(8)))  float  floatx8;

#define D_IN   4096
#define D_OUT  4096
#define BM     128
#define BN     128
#define BK     64          // two 32-wide WMMA K-steps per staged tile
#define LDA    72          // padded LDS row stride (elements): 144 B, 16 B aligned
#define LDB    72
#define LORA_SCALING 2.0f  // ALPHA / R = 16 / 8

__device__ __constant__ float c_nf4[16] = {
    -1.0f, -0.6962f, -0.5251f, -0.3949f, -0.2844f, -0.1848f, -0.0911f, 0.0f,
     0.0796f, 0.1609f, 0.2461f, 0.3379f, 0.4407f, 0.5626f, 0.723f, 1.0f};

// ---------------------------------------------------------------------------
// Kernel 1: hidden[M][8] = x[M][K] @ lora_a[K][8]   (fp32, per reference)
// One wave (32 lanes) per row; shuffle reduction.
// ---------------------------------------------------------------------------
__global__ __launch_bounds__(256)
void lora_hidden_kernel(const float* __restrict__ x,
                        const float* __restrict__ lora_a,
                        float* __restrict__ hidden, int M)
{
    const int lane = threadIdx.x & 31;
    const int wid  = threadIdx.x >> 5;
    const int row  = blockIdx.x * 8 + wid;
    if (row >= M) return;

    float acc[8];
#pragma unroll
    for (int r = 0; r < 8; ++r) acc[r] = 0.0f;

    const float* xr = x + (size_t)row * D_IN;
    for (int k = lane; k < D_IN; k += 32) {
        const float xv = xr[k];
        const float4 a0 = ((const float4*)(lora_a + (size_t)k * 8))[0];
        const float4 a1 = ((const float4*)(lora_a + (size_t)k * 8))[1];
        acc[0] += xv * a0.x; acc[1] += xv * a0.y;
        acc[2] += xv * a0.z; acc[3] += xv * a0.w;
        acc[4] += xv * a1.x; acc[5] += xv * a1.y;
        acc[6] += xv * a1.z; acc[7] += xv * a1.w;
    }
#pragma unroll
    for (int off = 16; off > 0; off >>= 1) {
#pragma unroll
        for (int r = 0; r < 8; ++r) acc[r] += __shfl_down(acc[r], off, 32);
    }
    if (lane == 0) {
        float4 h0 = {acc[0], acc[1], acc[2], acc[3]};
        float4 h1 = {acc[4], acc[5], acc[6], acc[7]};
        ((float4*)(hidden + (size_t)row * 8))[0] = h0;
        ((float4*)(hidden + (size_t)row * 8))[1] = h1;
    }
}

// ---------------------------------------------------------------------------
// Kernel 2: fused NF4-dequant bf16 WMMA GEMM + bias + LoRA epilogue.
// Block = 128x128 tile, 256 threads = 8 waves (2x4), each wave = 64x32
// = 4x2 tiles of v_wmma_f32_16x16x32_bf16, 2 K-steps per staged BK=64 tile.
// ---------------------------------------------------------------------------
__global__ __launch_bounds__(256)
void qlora_wmma_kernel(const float* __restrict__ x,
                       const int*   __restrict__ qw,
                       const float* __restrict__ scale,
                       const float* __restrict__ bias,
                       const float* __restrict__ lora_b,
                       const float* __restrict__ hidden,
                       float* __restrict__ out)
{
    __shared__ __bf16 Atile[BM * LDA];      // [m][k], row-major, K-contiguous
    __shared__ __bf16 Btile[BN * LDB];      // [n][k], transposed -> K-contiguous
    __shared__ float  sHidden[BM * 8];      // LoRA hidden rows for this block
    __shared__ float  sLoraB[8 * BN];       // LoRA B columns for this block
    __shared__ float  sNF4[16];             // NF4 codebook

    const int tid   = threadIdx.x;
    const int nBase = blockIdx.x * BN;
    const int mBase = blockIdx.y * BM;

    // ---- one-time staging of LoRA operands + codebook into LDS -----------
    for (int i = tid; i < BM * 8; i += 256)
        sHidden[i] = hidden[(size_t)mBase * 8 + i];
    for (int i = tid; i < 8 * BN; i += 256)
        sLoraB[i] = lora_b[(size_t)(i >> 7) * D_OUT + nBase + (i & 127)];
    if (tid < 16) sNF4[tid] = c_nf4[tid];

    const int lane = tid & 31;
    const int wid  = tid >> 5;
    const int wm   = wid >> 2;   // 0..1 -> 64 rows
    const int wn   = wid & 3;    // 0..3 -> 32 cols

    floatx8 acc[4][2];
#pragma unroll
    for (int i = 0; i < 4; ++i)
#pragma unroll
        for (int j = 0; j < 2; ++j)
            acc[i][j] = (floatx8){0.f, 0.f, 0.f, 0.f, 0.f, 0.f, 0.f, 0.f};

    // staging work assignment
    const int am  = tid >> 1;          // A: row 0..127
    const int ak  = (tid & 1) * 32;    // A: k half 0/32
    const int bk0 = (tid >> 3) * 2;    // B: k pair 0,2,..,62
    const int bn0 = (tid & 7) * 16;    // B: 16 consecutive n

    for (int kt = 0; kt < D_IN; kt += BK) {
        __syncthreads();   // protect LDS reuse (also orders initial staging)

        // ---- stage A: x fp32 -> bf16, [128][64] -------------------------
        {
            const float* gx = x + (size_t)(mBase + am) * D_IN + kt + ak;
            union { __bf16 h[32]; uint4 q[4]; } p;
#pragma unroll
            for (int i = 0; i < 8; ++i) {
                float4 f = ((const float4*)gx)[i];
                p.h[i * 4 + 0] = (__bf16)f.x;
                p.h[i * 4 + 1] = (__bf16)f.y;
                p.h[i * 4 + 2] = (__bf16)f.z;
                p.h[i * 4 + 3] = (__bf16)f.w;
            }
#pragma unroll
            for (int j = 0; j < 4; ++j)
                *((uint4*)&Atile[am * LDA + ak + j * 8]) = p.q[j];
            __builtin_prefetch(gx + BK, 0, 0);             // global_prefetch_b8
        }

        // ---- stage B: NF4 dequant -> bf16, stored [n][k] (transposed),
        //      two adjacent K values packed per dword -> ds_store_b32 ------
        {
            const int g = kt >> 7;  // 128-wide quant group, uniform per tile
            const int*   gq0 = qw    + (size_t)(kt + bk0) * D_OUT + nBase + bn0;
            const int*   gq1 = gq0 + D_OUT;
            const float* gs  = scale + (size_t)g * D_OUT + nBase + bn0;

            auto pack2 = [&](int qa, int qb, float s, int n) {
                union { __bf16 h[2]; unsigned u; } p;
                p.h[0] = (__bf16)(sNF4[qa + 8] * s);
                p.h[1] = (__bf16)(sNF4[qb + 8] * s);
                *((unsigned*)&Btile[n * LDB + bk0]) = p.u;
            };
#pragma unroll
            for (int c = 0; c < 4; ++c) {
                int4   qa = ((const int4*)gq0)[c];
                int4   qb = ((const int4*)gq1)[c];
                float4 s4 = ((const float4*)gs)[c];
                pack2(qa.x, qb.x, s4.x, bn0 + c * 4 + 0);
                pack2(qa.y, qb.y, s4.y, bn0 + c * 4 + 1);
                pack2(qa.z, qb.z, s4.z, bn0 + c * 4 + 2);
                pack2(qa.w, qb.w, s4.w, bn0 + c * 4 + 3);
            }
            __builtin_prefetch(gq0 + (size_t)BK * D_OUT, 0, 0);
        }
        __syncthreads();

        // ---- fragments (ISA §7.12.2 layouts) + 2 x 8 WMMAs --------------
        const int fm = lane & 15;
        const int kh = lane >> 4;
        union Frag { bf16x16 v; uint4 q[2]; };
#pragma unroll
        for (int ks = 0; ks < 2; ++ks) {
            Frag a[4], b[2];
#pragma unroll
            for (int tm = 0; tm < 4; ++tm) {
                // A lane: row fm, K = ks*32 + {kh*8..+7} and {kh*8+16..+23}
                const __bf16* p =
                    &Atile[(wm * 64 + tm * 16 + fm) * LDA + ks * 32 + kh * 8];
                a[tm].q[0] = *((const uint4*)p);
                a[tm].q[1] = *((const uint4*)(p + 16));
            }
#pragma unroll
            for (int tn = 0; tn < 2; ++tn) {
                // B lane: col fm, K = ks*32 + {kh*8..+7} and {kh*8+16..+23}
                const __bf16* p =
                    &Btile[(wn * 32 + tn * 16 + fm) * LDB + ks * 32 + kh * 8];
                b[tn].q[0] = *((const uint4*)p);
                b[tn].q[1] = *((const uint4*)(p + 16));
            }
#pragma unroll
            for (int tm = 0; tm < 4; ++tm)
#pragma unroll
                for (int tn = 0; tn < 2; ++tn)
                    acc[tm][tn] = __builtin_amdgcn_wmma_f32_16x16x32_bf16(
                        false, a[tm].v, false, b[tn].v,
                        (short)0, acc[tm][tn], false, false);
        }
    }

    // ---- epilogue: bf16 rounding of (acc + bias) + fp32 LoRA delta ------
    const int fm   = lane & 15;
    const int half = lane >> 4;
#pragma unroll
    for (int tn = 0; tn < 2; ++tn) {
        const int jl = wn * 32 + tn * 16 + fm;
        const float bsc = (float)(__bf16)bias[nBase + jl];
#pragma unroll
        for (int tm = 0; tm < 4; ++tm) {
#pragma unroll
            for (int v = 0; v < 8; ++v) {
                const int il = wm * 64 + tm * 16 + half * 8 + v;  // C/D layout
                float delta = 0.f;
#pragma unroll
                for (int r = 0; r < 8; ++r)
                    delta += sHidden[il * 8 + r] * sLoraB[r * 128 + jl];
                const float basev = (float)(__bf16)(acc[tm][tn][v] + bsc);
                out[(size_t)(mBase + il) * D_OUT + nBase + jl] =
                    basev + delta * LORA_SCALING;
            }
        }
    }
}

// ---------------------------------------------------------------------------
// launch: inputs in setup_inputs() order: x, qweight, scale, bias, lora_a, lora_b
// ---------------------------------------------------------------------------
extern "C" void kernel_launch(void* const* d_in, const int* in_sizes, int n_in,
                              void* d_out, int out_size, void* d_ws, size_t ws_size,
                              hipStream_t stream)
{
    const float* x      = (const float*)d_in[0];
    const int*   qw     = (const int*)  d_in[1];
    const float* scale  = (const float*)d_in[2];
    const float* bias   = (const float*)d_in[3];
    const float* lora_a = (const float*)d_in[4];
    const float* lora_b = (const float*)d_in[5];
    float*       out    = (float*)d_out;

    const int M = in_sizes[0] / D_IN;          // 8192
    float* hidden = (float*)d_ws;              // M*8 fp32 = 256 KB scratch

    lora_hidden_kernel<<<(M + 7) / 8, 256, 0, stream>>>(x, lora_a, hidden, M);

    dim3 grid(D_OUT / BN, M / BM);             // 32 x 64 = 2048 blocks
    qlora_wmma_kernel<<<grid, 256, 0, stream>>>(x, qw, scale, bias,
                                                lora_b, hidden, out);
}
